// WindowAttention_45981919871211
// MI455X (gfx1250) — matrix operand order repaired and checked by reading, not compiled, not run on hardware
//
#include <hip/hip_runtime.h>
#include <hip/hip_bf16.h>
#include <math.h>

#define NTOK 49
#define CDIM 256
#define HEADS 8
#define HD 32
#define QKVN 768
#define NPAD 64
#define NWIN 64   // mask.shape[0]

// workspace byte offsets
#define WS_SCALE_OFF 0              // 8 floats
#define WS_BIAS_OFF  32             // 8*49*49 floats  (ends 76864)
#define WS_WQ_OFF    76928          // 768*256 f16     (ends 470144)
#define WS_WP_OFF    470144         // 256*256 f16     (ends 601216)

typedef __attribute__((ext_vector_type(16))) _Float16 v16h;
typedef __attribute__((ext_vector_type(8)))  float    v8f;
typedef __attribute__((ext_vector_type(4)))  float    f32x4;
typedef __attribute__((ext_vector_type(4)))  _Float16 f16x4;

// type expected by the async-to-LDS builtin (per hipcc diagnostic)
typedef int v4i __attribute__((vector_size(4 * sizeof(int))));
typedef __attribute__((address_space(3))) v4i lds_v4i;

union AFrag { v16h v; _Float16 e[16]; };
union CFrag { v8f  v; float    e[8];  };

#if defined(__has_builtin)
#if __has_builtin(__builtin_amdgcn_global_load_async_to_lds_b128)
#define HAVE_ASYNC_LDS 1
#endif
#endif

__device__ inline CFrag zero8() {
    CFrag c;
#pragma unroll
    for (int r = 0; r < 8; ++r) c.e[r] = 0.f;
    return c;
}

// ---- A fragment (16x32 f16) from LDS f16 matrix with given row stride
__device__ inline AFrag load_a_lds(const _Float16* base, int stride, int mb, int kbase, int lane) {
    AFrag a;
    const _Float16* p = base + (mb + (lane & 15)) * stride + kbase + ((lane >> 4) << 4);
#pragma unroll
    for (int j = 0; j < 16; ++j) a.e[j] = p[j];
    return a;
}

// ---- B fragment (32x16 f16) from global f16 weight W[N][K] row-major (B = W^T)
__device__ inline AFrag load_b_wh(const _Float16* __restrict__ W, int ldk, int nb, int kb, int lane) {
    AFrag bm;
    const _Float16* p = W + (size_t)(nb + (lane & 15)) * ldk + kb + ((lane >> 4) << 4);
#pragma unroll
    for (int j = 0; j < 16; ++j) bm.e[j] = p[j];
    return bm;
}

// ---- B fragment for K^T: column = token m, K dim = d (contiguous in LDS q/k row)
__device__ inline AFrag load_b_kt(const _Float16* qkv, int h, int nb, int lane) {
    AFrag bm;
    const _Float16* p = qkv + (nb + (lane & 15)) * QKVN + 256 + h * HD + ((lane >> 4) << 4);
#pragma unroll
    for (int j = 0; j < 16; ++j) bm.e[j] = p[j];
    return bm;
}

// ---- B fragment for V: column = d, K dim = token m (stride QKVN in LDS)
__device__ inline AFrag load_b_v(const _Float16* qkv, int h, int db, int kb, int lane) {
    AFrag bm;
    int d = db + (lane & 15);
    const _Float16* p = qkv + (size_t)(kb + ((lane >> 4) << 4)) * QKVN + 512 + h * HD + d;
#pragma unroll
    for (int j = 0; j < 16; ++j) bm.e[j] = p[(size_t)j * QKVN];
    return bm;
}

// =====================================================================
// Kernel 0: CPB MLP -> 16*sigmoid(bias) table, head scales, f16 weights
// =====================================================================
__global__ __launch_bounds__(256) void setup_kernel(
    const float* __restrict__ rct,         // [169,2]
    const float* __restrict__ w1,          // [512,2]
    const float* __restrict__ b1,          // [512]
    const float* __restrict__ w2,          // [8,512]
    const float* __restrict__ logit_scale, // [8]
    const int*   __restrict__ rpi,         // [49*49]
    const float* __restrict__ qkv_w,       // [768,256]
    const float* __restrict__ proj_w,      // [256,256]
    char* __restrict__ wsb)
{
    float*     wsS  = (float*)(wsb + WS_SCALE_OFF);
    float*     wsB  = (float*)(wsb + WS_BIAS_OFF);
    _Float16*  wq   = (_Float16*)(wsb + WS_WQ_OFF);
    _Float16*  wp   = (_Float16*)(wsb + WS_WP_OFF);

    __shared__ float table8[169 * 8];
    int tid = threadIdx.x;
    if (tid < 8) wsS[tid] = __expf(fminf(logit_scale[tid], 4.6051702f)); // ln(100)

    // weight fp32 -> f16 conversion (done once; halves per-block L2 traffic)
    for (int i = tid; i < QKVN * CDIM; i += 256) wq[i] = (_Float16)qkv_w[i];
    for (int i = tid; i < CDIM * CDIM; i += 256) wp[i] = (_Float16)proj_w[i];

    for (int p = tid; p < 169; p += 256) {
        float t0 = rct[p * 2], t1 = rct[p * 2 + 1];
        float acc[8];
#pragma unroll
        for (int h = 0; h < 8; ++h) acc[h] = 0.f;
        for (int j = 0; j < 512; ++j) {
            float hv = fmaxf(t0 * w1[j * 2] + t1 * w1[j * 2 + 1] + b1[j], 0.f);
#pragma unroll
            for (int h = 0; h < 8; ++h) acc[h] += hv * w2[h * 512 + j];
        }
#pragma unroll
        for (int h = 0; h < 8; ++h) table8[p * 8 + h] = acc[h];
    }
    __syncthreads();
    for (int i = tid; i < HEADS * NTOK * NTOK; i += 256) {
        int h = i / (NTOK * NTOK);
        int rem = i - h * (NTOK * NTOK);
        float bv = table8[rpi[rem] * 8 + h];
        wsB[i] = 16.f / (1.f + __expf(-bv));
    }
}

// =====================================================================
// Kernel 1: fully fused window attention (one block per window)
// Dynamic LDS (288 KB):
//   [0, 96K)        qkv f16 [64][768]
//   [96K, 160K)     P   f16 [8][64][64]
//   [160K, 288K)    S   f32 [8][64][64]
//     front  64KB reused first as staged x fp32 [64][256], later attn_out f16
//     second 64KB reused first as x f16 image [64][256] (32KB)
// =====================================================================
__global__ __launch_bounds__(256) void attn_kernel(
    const float* __restrict__ x,
    const float* __restrict__ mask,
    const float* __restrict__ q_bias,
    const float* __restrict__ v_bias,
    const float* __restrict__ proj_b,
    const char*  __restrict__ wsb,
    float* __restrict__ out)
{
    extern __shared__ char smem[];
    _Float16* sQKV = (_Float16*)smem;                  // 64*768 halves
    _Float16* sP   = (_Float16*)(smem + 98304);        // 8*64*64 halves
    float*    sS   = (float*)(smem + 163840);          // 8*64*64 floats
    float*    sX   = (float*)(smem + 163840);          // 64*256 floats (overlay, 64KB)
    _Float16* sXh  = (_Float16*)(smem + 163840 + 65536); // 64*256 halves (overlay, 32KB)
    _Float16* sAO  = (_Float16*)(smem + 163840);       // 64*256 halves (overlay)

    const float*    wsS = (const float*)(wsb + WS_SCALE_OFF);
    const float*    wsB = (const float*)(wsb + WS_BIAS_OFF);
    const _Float16* wq  = (const _Float16*)(wsb + WS_WQ_OFF);
    const _Float16* wp  = (const _Float16*)(wsb + WS_WP_OFF);

    const int b    = blockIdx.x;
    const int tid  = threadIdx.x;
    const int lane = tid & 31;
    const int wave = tid >> 5;
    const float* xb = x + (size_t)b * NTOK * CDIM;

    // ---------------- Phase 0a: stage x into LDS (async DMA path), zero pad rows,
    // and prefetch this wave's qkv weight slab into the cache hierarchy.
    {
        const _Float16* wslab = wq + (size_t)wave * 96 * CDIM;
#pragma unroll 1
        for (int off = lane * 64; off < 96 * CDIM; off += 32 * 64)
            __builtin_prefetch(wslab + off, 0, 0);   // -> global_prefetch_b8

#ifdef HAVE_ASYNC_LDS
#pragma unroll 1
        for (int i = tid; i < (NTOK * CDIM) / 4; i += 256) {
            v4i* gp = (v4i*)(xb + (size_t)i * 4);
            lds_v4i* lp = (lds_v4i*)((char*)sX + (size_t)i * 16);
            __builtin_amdgcn_global_load_async_to_lds_b128(gp, lp, 0, 0);
        }
#else
#pragma unroll 1
        for (int i = tid; i < (NTOK * CDIM) / 4; i += 256)
            ((f32x4*)sX)[i] = ((const f32x4*)xb)[i];
#endif
        // zero padding rows 49..63 (fp32 image)
        f32x4 z; z.x = 0.f; z.y = 0.f; z.z = 0.f; z.w = 0.f;
#pragma unroll 1
        for (int i = tid; i < ((NPAD - NTOK) * CDIM) / 4; i += 256)
            ((f32x4*)sX)[(NTOK * CDIM) / 4 + i] = z;
#ifdef HAVE_ASYNC_LDS
        asm volatile("s_wait_asynccnt 0x0" ::: "memory");
#endif
    }
    __syncthreads();

    // ---------------- Phase 0b: one-shot fp32 -> f16 conversion of the x image
    // (so the hot Phase-1 A loads are pure b128 LDS reads, no per-fragment cvt)
#pragma unroll 1
    for (int i = tid; i < (NPAD * CDIM) / 4; i += 256) {
        f32x4 vsrc = ((const f32x4*)sX)[i];
        f16x4 vdst;
        vdst.x = (_Float16)vsrc.x; vdst.y = (_Float16)vsrc.y;
        vdst.z = (_Float16)vsrc.z; vdst.w = (_Float16)vsrc.w;
        ((f16x4*)sXh)[i] = vdst;
    }
    __syncthreads();

    // ---------------- Phase 1: QKV = x @ qkv_w^T + bias  (64x768, f16 WMMA)
#pragma unroll 1
    for (int nt = 0; nt < 6; ++nt) {
        int nb   = wave * 96 + nt * 16;
        int colg = nb + (lane & 15);
        float bias = (colg < 256) ? q_bias[colg]
                   : ((colg < 512) ? 0.f : v_bias[colg - 512]);
#pragma unroll 1
        for (int mt = 0; mt < 4; ++mt) {
            int mb = mt * 16;
            CFrag c = zero8();
#pragma unroll
            for (int kt = 0; kt < 8; ++kt) {
                AFrag a  = load_a_lds(sXh, CDIM, mb, kt * 32, lane);
                AFrag bm = load_b_wh(wq, CDIM, nb, kt * 32, lane);
                c.v = __builtin_amdgcn_wmma_f32_16x16x32_f16(
                        false, a.v, false, bm.v, (short)0, c.v, false, false);
            }
            int rb = mb + ((lane >> 4) << 3);
#pragma unroll
            for (int r = 0; r < 8; ++r)
                sQKV[(rb + r) * QKVN + colg] = (_Float16)(c.e[r] + bias);
        }
    }
    __syncthreads();

    // ---------------- Phase 2: l2-normalize q,k rows; fold head scale into q
    for (int g = tid; g < NPAD * 16; g += 256) {
        int t = g >> 4; int rem = g & 15;
        int sqk = rem & 1; int h2 = rem >> 1;
        _Float16* p = sQKV + t * QKVN + sqk * 256 + h2 * HD;
        float ssum = 0.f;
#pragma unroll
        for (int d = 0; d < HD; ++d) { float vv = (float)p[d]; ssum += vv * vv; }
        float sc = 1.f / fmaxf(sqrtf(ssum), 1e-12f);
        if (sqk == 0) sc *= wsS[h2];
#pragma unroll
        for (int d = 0; d < HD; ++d) p[d] = (_Float16)((float)p[d] * sc);
    }
    __syncthreads();

    // ---------------- Phase 3: S = scale*(q_n @ k_n^T) + 16*sigmoid(bias) + mask
    const int h = wave;  // one wave per head
    const float* biasW = wsB + h * (NTOK * NTOK);
    const float* maskW = mask + (size_t)(b & (NWIN - 1)) * (NTOK * NTOK);
#pragma unroll 1
    for (int mt = 0; mt < 4; ++mt) {
#pragma unroll 1
        for (int nt = 0; nt < 4; ++nt) {
            CFrag c = zero8();
            AFrag a  = load_a_lds(sQKV, QKVN, mt * 16, h * HD, lane); // q slice
            AFrag bm = load_b_kt(sQKV, h, nt * 16, lane);             // k^T slice
            c.v = __builtin_amdgcn_wmma_f32_16x16x32_f16(
                    false, a.v, false, bm.v, (short)0, c.v, false, false);
            int m  = nt * 16 + (lane & 15);
            int rb = mt * 16 + ((lane >> 4) << 3);
#pragma unroll
            for (int r = 0; r < 8; ++r) {
                int n = rb + r;
                float val = c.e[r];
                if (m < NTOK) {
                    if (n < NTOK) val += biasW[n * NTOK + m] + maskW[n * NTOK + m];
                } else {
                    val = -1e30f;  // mask out padded key tokens
                }
                sS[h * 4096 + n * 64 + m] = val;
            }
        }
    }
    // same-wave LDS RAW: DScnt ordering handles this; no block barrier needed

    // ---------------- Phase 4: row softmax -> P (f16)
    {
        float*    Sh = sS + h * 4096;
        _Float16* Ph = sP + h * 4096;
#pragma unroll 1
        for (int rr = 0; rr < 2; ++rr) {
            int n = lane + rr * 32;
            const float* srow = Sh + n * 64;
            float mx = -1e30f;
            for (int m2 = 0; m2 < 64; ++m2) mx = fmaxf(mx, srow[m2]);
            float ssum = 0.f;
            for (int m2 = 0; m2 < 64; ++m2) ssum += __expf(srow[m2] - mx);
            float inv = 1.f / ssum;
            for (int m2 = 0; m2 < 64; ++m2)
                Ph[n * 64 + m2] = (_Float16)(__expf(srow[m2] - mx) * inv);
        }
    }
    __syncthreads();  // all waves done reading S before region is reused as attn_out

    // ---------------- Phase 5: O = P @ V  -> attn_out f16 [64][256]
#pragma unroll 1
    for (int mt = 0; mt < 4; ++mt) {
#pragma unroll 1
        for (int dt = 0; dt < 2; ++dt) {
            CFrag c = zero8();
#pragma unroll
            for (int kt = 0; kt < 2; ++kt) {
                AFrag a  = load_a_lds(sP + h * 4096, 64, mt * 16, kt * 32, lane);
                AFrag bm = load_b_v(sQKV, h, dt * 16, kt * 32, lane);
                c.v = __builtin_amdgcn_wmma_f32_16x16x32_f16(
                        false, a.v, false, bm.v, (short)0, c.v, false, false);
            }
            int d  = dt * 16 + (lane & 15);
            int rb = mt * 16 + ((lane >> 4) << 3);
#pragma unroll
            for (int r = 0; r < 8; ++r)
                sAO[(rb + r) * CDIM + h * HD + d] = (_Float16)c.e[r];
        }
    }
    __syncthreads();

    // ---------------- Phase 6: Y = attn_out @ proj_w^T + proj_b -> global
#pragma unroll 1
    for (int nt = 0; nt < 2; ++nt) {
        int nb   = wave * 32 + nt * 16;
        int colg = nb + (lane & 15);
        float pb = proj_b[colg];
#pragma unroll 1
        for (int mt = 0; mt < 4; ++mt) {
            CFrag c = zero8();
#pragma unroll
            for (int kt = 0; kt < 8; ++kt) {
                AFrag a  = load_a_lds(sAO, CDIM, mt * 16, kt * 32, lane);
                AFrag bm = load_b_wh(wp, CDIM, nb, kt * 32, lane);
                c.v = __builtin_amdgcn_wmma_f32_16x16x32_f16(
                        false, a.v, false, bm.v, (short)0, c.v, false, false);
            }
            int rb = mt * 16 + ((lane >> 4) << 3);
#pragma unroll
            for (int r = 0; r < 8; ++r) {
                int n = rb + r;
                if (n < NTOK)
                    out[((size_t)b * NTOK + n) * CDIM + colg] = c.e[r] + pb;
            }
        }
    }
}

extern "C" void kernel_launch(void* const* d_in, const int* in_sizes, int n_in,
                              void* d_out, int out_size, void* d_ws, size_t ws_size,
                              hipStream_t stream) {
    (void)in_sizes; (void)n_in; (void)out_size; (void)ws_size;
    const float* x           = (const float*)d_in[0];
    const float* mask        = (const float*)d_in[1];
    const float* qkv_w       = (const float*)d_in[2];
    const float* q_bias      = (const float*)d_in[3];
    const float* v_bias      = (const float*)d_in[4];
    const float* logit_scale = (const float*)d_in[5];
    const float* cpb_w1      = (const float*)d_in[6];
    const float* cpb_b1      = (const float*)d_in[7];
    const float* cpb_w2      = (const float*)d_in[8];
    const float* proj_w      = (const float*)d_in[9];
    const float* proj_b      = (const float*)d_in[10];
    const float* rct         = (const float*)d_in[11];
    const int*   rpi         = (const int*)d_in[12];
    float* out = (float*)d_out;
    char*  wsb = (char*)d_ws;

    const size_t shmem = 294912;  // 288 KB < 320 KB per-WGP LDS
    (void)hipFuncSetAttribute((const void*)attn_kernel,
                              hipFuncAttributeMaxDynamicSharedMemorySize, (int)shmem);

    setup_kernel<<<1, 256, 0, stream>>>(rct, cpb_w1, cpb_b1, cpb_w2, logit_scale,
                                        rpi, qkv_w, proj_w, wsb);
    attn_kernel<<<2048, 256, shmem, stream>>>(x, mask, q_bias, v_bias, proj_b, wsb, out);
}
